// MultiheadSelfAttention_17463337026245
// MI455X (gfx1250) — matrix operand order repaired and checked by reading, not compiled, not run on hardware
//
#include <hip/hip_runtime.h>
#include <cstdint>

#define B_ 2
#define H_ 8
#define T_ 4096
#define D_ 256
#define R_ 32

typedef __attribute__((ext_vector_type(16))) __bf16    v16bf;
typedef __attribute__((ext_vector_type(8)))  float     v8f;
typedef __attribute__((ext_vector_type(8)))  uint32_t  v8u;
typedef __attribute__((ext_vector_type(4)))  unsigned int u32x4;
typedef __attribute__((ext_vector_type(8)))  int       i32x8;
typedef __attribute__((ext_vector_type(4)))  int       i32x4;

union FragAB { v16bf bf; v8u u; };

static __device__ __forceinline__ v8f wmma_bf16(FragAB a, FragAB b, v8f c) {
  return __builtin_amdgcn_wmma_f32_16x16x32_bf16(false, a.bf, false, b.bf,
                                                 (short)0, c, false, false);
}

// f32 -> bf16 round-to-nearest-even (bit pattern as ushort)
static __device__ __forceinline__ unsigned short f2bf(float f) {
  uint32_t u = __float_as_uint(f);
  u += 0x7FFFu + ((u >> 16) & 1u);
  return (unsigned short)(u >> 16);
}

static __device__ __forceinline__ uint32_t ld2(const unsigned short* p) {
  return *(const uint32_t*)p;   // two packed bf16 (lo = first)
}

// ---- Tensor Data Mover: 1D contiguous tile (n2 bf16 elements) global -> LDS.
// D# group0: count=1, lds_addr, global_addr[56:0], type=2.
// D# group1: data_size=2B; tensor_dim0=n2; tensor_dim1=1; tile_dim0=n2;
//            tile_dim1=0 (unused -> 1D); dim0_stride=n2.
static __device__ __forceinline__ void tdm_load_1d(uint32_t lds_off,
                                                   const unsigned short* gptr,
                                                   uint32_t n2) {
  uint64_t ga = (uint64_t)(uintptr_t)gptr;
  u32x4 g0;
  g0.x = 1u;                                   // count=1, user descriptor
  g0.y = lds_off;                              // LDS byte address
  g0.z = (uint32_t)ga;                         // global addr lo
  g0.w = ((uint32_t)(ga >> 32) & 0x01FFFFFFu) | (2u << 30);  // hi | type=2
  i32x8 g1;
  g1[0] = (int)(1u << 16);                     // data_size = 1 (2 bytes)
  g1[1] = (int)((n2 & 0xFFFFu) << 16);         // tensor_dim0 lo16
  g1[2] = (int)(((n2 >> 16) & 0xFFFFu) | (1u << 16)); // dim0 hi16 | dim1 lo16=1
  g1[3] = (int)((n2 & 0xFFFFu) << 16);         // dim1 hi16=0 | tile_dim0
  g1[4] = 0;                                   // tile_dim1=0, tile_dim2=0
  g1[5] = (int)n2;                             // tensor_dim0_stride lo32
  g1[6] = 0;                                   // stride0 hi | stride1 lo
  g1[7] = 0;
  i32x4 z4 = {0, 0, 0, 0};
  i32x8 z8 = {0, 0, 0, 0, 0, 0, 0, 0};
  __builtin_amdgcn_tensor_load_to_lds(g0, g1, z4, z4, z8, 0);
}

static __device__ __forceinline__ uint32_t lds_off_of(const void* p) {
  return (uint32_t)(uintptr_t)p;   // low 32 bits of generic LDS ptr = offset
}

// ---------------------------------------------------------------- convert
__global__ void k_convert(const float* __restrict__ src,
                          unsigned short* __restrict__ dst, int n) {
  int i = blockIdx.x * blockDim.x + threadIdx.x;
  int stride = gridDim.x * blockDim.x;
  for (; i < n; i += stride) dst[i] = f2bf(src[i]);
}

// ---------------------------------------------------------------- q/k projection
// q[b,h,t,r] = (x[b,t,:] . Q[h,r,:]) / 16 ; k analogous (no scale)
// grid = B * (T/64) * H, block = 256 (8 waves)
__global__ __launch_bounds__(256) void k_proj(
    const unsigned short* __restrict__ xb,
    const unsigned short* __restrict__ Qb,
    const unsigned short* __restrict__ Kb,
    unsigned short* __restrict__ q_ws,
    unsigned short* __restrict__ k_ws) {
  __shared__ __align__(16) unsigned short x_lds[64 * D_];
  int bx = blockIdx.x;
  int h = bx & 7, tb = (bx >> 3) & 63, b = bx >> 9;
  int tbase = tb * 64;
  int tid = threadIdx.x;

  {
    const uint4* src = (const uint4*)(xb + ((size_t)b * T_ + tbase) * D_);
    uint4* dst = (uint4*)x_lds;
    for (int i = tid; i < 2048; i += 256) dst[i] = src[i];
  }
  __syncthreads();

  int w = tid >> 5, lane = tid & 31;
  int rb = w & 3;
  bool isQ = (w < 4);
  const unsigned short* mat = (isQ ? Qb : Kb) + (size_t)h * R_ * D_;
  int ln = lane & 15, kh = lane >> 4;
  int m = rb * 16 + ln;
  v8f acc0 = {}, acc1 = {};

  for (int kk = 0; kk < D_; kk += 32) {
    FragAB a;
#pragma unroll
    for (int j = 0; j < 4; ++j) {
      a.u[j]     = ld2(&x_lds[m * D_ + kk + kh * 8 + 2 * j]);
      a.u[j + 4] = ld2(&x_lds[m * D_ + kk + 16 + kh * 8 + 2 * j]);
    }
#pragma unroll
    for (int ct = 0; ct < 2; ++ct) {
      int r = ct * 16 + ln;
      FragAB bf;
#pragma unroll
      for (int j = 0; j < 8; ++j)
        bf.u[j] = ld2(&mat[(size_t)r * D_ + kk + kh * 8 + 2 * j + (j >= 4 ? 8 : 0)]);
      if (ct == 0) acc0 = wmma_bf16(a, bf, acc0);
      else         acc1 = wmma_bf16(a, bf, acc1);
    }
  }
  float scale = isQ ? 0.0625f : 1.0f;   // 1/sqrt(256) folded into q
  unsigned short* outp =
      (isQ ? q_ws : k_ws) + (((size_t)b * H_ + h) * T_ + tbase) * R_;
#pragma unroll
  for (int v = 0; v < 8; ++v) {
    int row = rb * 16 + kh * 8 + v;
    outp[(size_t)row * R_ + ln]      = f2bf(acc0[v] * scale);
    outp[(size_t)row * R_ + 16 + ln] = f2bf(acc1[v] * scale);
  }
}

// ---------------------------------------------------------------- flash attention
// grid = B * H * (T/64), block = 256 (8 waves)
// TDM double-buffered staging of k chunks (32x32) and x chunks (32x256).
__global__ __launch_bounds__(256) void k_flash(
    const unsigned short* __restrict__ q_ws,
    const unsigned short* __restrict__ k_ws,
    const unsigned short* __restrict__ xb,
    unsigned short* __restrict__ ctx_ws) {
  __shared__ __align__(16) unsigned short q_lds[64 * R_];
  __shared__ __align__(16) unsigned short k_lds[2][32 * R_];
  __shared__ __align__(16) unsigned short xs_lds[2][32 * D_];  // staging [u][d]
  __shared__ __align__(16) unsigned short xT_lds[D_ * 32];     // [d][u]
  __shared__ __align__(16) unsigned short p_lds[64 * 32];      // [t][u]
  __shared__ float red_lds[8 * 32 * 8];
  __shared__ float rinv_lds[64];

  int bx = blockIdx.x;
  int tq = bx & 63, h = (bx >> 6) & 7, b = bx >> 9;
  int tbase = tq * 64;
  int tid = threadIdx.x;
  int w = tid >> 5, lane = tid & 31;
  int rb = w & 3, cb = w >> 2;
  int ln = lane & 15, kh = lane >> 4;

  const unsigned short* qptr =
      q_ws + (((size_t)b * H_ + h) * T_ + tbase) * R_;
  const unsigned short* kptr = k_ws + (((size_t)b * H_ + h) * T_) * R_;
  const unsigned short* xptr = xb + (size_t)b * T_ * D_;

  // prologue: TDM q tile + first k/x chunks into buffer 0
  if (w == 0) {
    tdm_load_1d(lds_off_of(q_lds), qptr, 64 * R_);
    tdm_load_1d(lds_off_of(&k_lds[0][0]), kptr, 32 * R_);
    tdm_load_1d(lds_off_of(&xs_lds[0][0]), xptr, 32 * D_);
    __builtin_amdgcn_s_wait_tensorcnt(0);
  }
  __syncthreads();

  FragAB aq;   // persistent A-fragment of q
  {
    int m = rb * 16 + ln;
#pragma unroll
    for (int j = 0; j < 4; ++j) {
      aq.u[j]     = ld2(&q_lds[m * R_ + kh * 8 + 2 * j]);
      aq.u[j + 4] = ld2(&q_lds[m * R_ + 16 + kh * 8 + 2 * j]);
    }
  }

  v8f acc[8];
#pragma unroll
  for (int i = 0; i < 8; ++i) acc[i] = {};
  float rs[8] = {0, 0, 0, 0, 0, 0, 0, 0};

  const int NIT = T_ / 32;
  for (int it = 0; it < NIT; ++it) {
    int cur = it & 1;
    // prefetch next chunk via TDM while we compute on current buffers
    if (w == 0 && (it + 1) < NIT) {
      int ubn = (it + 1) * 32;
      tdm_load_1d(lds_off_of(&k_lds[cur ^ 1][0]),
                  kptr + (size_t)ubn * R_, 32 * R_);
      tdm_load_1d(lds_off_of(&xs_lds[cur ^ 1][0]),
                  xptr + (size_t)ubn * D_, 32 * D_);
    }

    // K-transpose x chunk LDS->LDS: xs [u][d] -> xT [d][u]
    {
      int u = tid >> 3;
      int dbase = (tid & 7) * 32;
      const uint4* row = (const uint4*)&xs_lds[cur][u * D_ + dbase];
      unsigned short tmp[32];
      *(uint4*)&tmp[0]  = row[0];
      *(uint4*)&tmp[8]  = row[1];
      *(uint4*)&tmp[16] = row[2];
      *(uint4*)&tmp[24] = row[3];
#pragma unroll
      for (int i = 0; i < 32; ++i) xT_lds[(dbase + i) * 32 + u] = tmp[i];
    }
    __syncthreads();

    // S tile = q (16x32) x k^T (32x16) -- one WMMA per wave
    FragAB kb;
    {
      int ucol = cb * 16 + ln;
#pragma unroll
      for (int j = 0; j < 8; ++j)
        kb.u[j] = ld2(&k_lds[cur][ucol * R_ + kh * 8 + 2 * j + (j >= 4 ? 8 : 0)]);
    }
    v8f zc = {};
    v8f s = wmma_bf16(aq, kb, zc);

    // exp (max-free softmax), stash P tile to LDS, accumulate row sums
#pragma unroll
    for (int v = 0; v < 8; ++v) {
      float p = __expf(s[v]);
      rs[v] += p;
      int row = rb * 16 + kh * 8 + v;
      p_lds[row * 32 + cb * 16 + ln] = f2bf(p);
    }
    __syncthreads();

    // ctx[64x256] += P[64x32] x Xchunk[32x256] -- 8 WMMAs per wave
    FragAB ap;
    {
      int m = rb * 16 + ln;
#pragma unroll
      for (int j = 0; j < 4; ++j) {
        ap.u[j]     = ld2(&p_lds[m * 32 + kh * 8 + 2 * j]);
        ap.u[j + 4] = ld2(&p_lds[m * 32 + 16 + kh * 8 + 2 * j]);
      }
    }
#pragma unroll
    for (int ct = 0; ct < 8; ++ct) {
      int d = cb * 128 + ct * 16 + ln;
      FragAB xf;
#pragma unroll
      for (int j = 0; j < 8; ++j)
        xf.u[j] = ld2(&xT_lds[d * 32 + kh * 8 + 2 * j + (j >= 4 ? 8 : 0)]);
      acc[ct] = wmma_bf16(ap, xf, acc[ct]);
    }
    // drain prefetch (wave 0 tracks TENSORcnt), then publish via barrier
    if (w == 0) __builtin_amdgcn_s_wait_tensorcnt(0);
    __syncthreads();
  }

  // deterministic fixed-order row-sum reduction
#pragma unroll
  for (int v = 0; v < 8; ++v) red_lds[(w * 32 + lane) * 8 + v] = rs[v];
  __syncthreads();
  if (tid < 64) {
    int row = tid;
    int rrb = row >> 4, r16 = row & 15;
    int hh = r16 >> 3, vv = r16 & 7;
    float s = 0.f;
#pragma unroll
    for (int c = 0; c < 2; ++c) {
      int wv = rrb + 4 * c;
      for (int l = 0; l < 16; ++l)
        s += red_lds[(wv * 32 + hh * 16 + l) * 8 + vv];
    }
    rinv_lds[row] = 1.0f / s;
  }
  __syncthreads();

  // normalize and store ctx as bf16, layout [b][t][h*256+d]
  unsigned short* ctxp =
      ctx_ws + ((size_t)b * T_ + tbase) * (H_ * D_) + h * D_;
#pragma unroll
  for (int v = 0; v < 8; ++v) {
    int row = rb * 16 + kh * 8 + v;
    float sc = rinv_lds[row];
#pragma unroll
    for (int ct = 0; ct < 8; ++ct) {
      int d = cb * 128 + ct * 16 + ln;
      ctxp[(size_t)row * (H_ * D_) + d] = f2bf(acc[ct][v] * sc);
    }
  }
}

// ---------------------------------------------------------------- VO projection
// out[b,t,e] = sum_{hd} ctx[b,t,hd] * VO[hd,e] ; grid = B*(T/64)*4, block=256
__global__ __launch_bounds__(256) void k_out(
    const unsigned short* __restrict__ ctx_ws,
    const unsigned short* __restrict__ VOb,
    float* __restrict__ out) {
  __shared__ __align__(16) unsigned short a_lds[64 * 32];
  __shared__ __align__(16) unsigned short voT_lds[64 * 32];  // [e][k]
  int bx = blockIdx.x;
  int cblk = bx & 3, tb = (bx >> 2) & 63, b = bx >> 8;
  int tbase = tb * 64, ebase = cblk * 64;
  int tid = threadIdx.x;
  int w = tid >> 5, lane = tid & 31;
  int rb = w & 3, cn = w >> 2;
  int ln = lane & 15, kh = lane >> 4;
  const int KD = H_ * D_;  // 2048
  v8f acc0 = {}, acc1 = {};

  for (int kk = 0; kk < KD; kk += 32) {
    {  // A: ctx rows tbase..+63, cols kk..+31
      int row = tid >> 2, seg = tid & 3;
      const uint4* src = (const uint4*)(ctx_ws +
          ((size_t)b * T_ + tbase + row) * KD + kk + seg * 8);
      ((uint4*)a_lds)[row * 4 + seg] = *src;
    }
    {  // B: VO rows kk..+31 x cols ebase..+63, transposed into [e][k]
      int kl = tid >> 3, eb = (tid & 7) * 8;
      const uint4* src =
          (const uint4*)(VOb + (size_t)(kk + kl) * D_ + ebase + eb);
      unsigned short tmp[8];
      *(uint4*)tmp = *src;
#pragma unroll
      for (int i = 0; i < 8; ++i) voT_lds[(eb + i) * 32 + kl] = tmp[i];
    }
    __syncthreads();
    FragAB a;
    {
      int m = rb * 16 + ln;
#pragma unroll
      for (int j = 0; j < 4; ++j) {
        a.u[j]     = ld2(&a_lds[m * 32 + kh * 8 + 2 * j]);
        a.u[j + 4] = ld2(&a_lds[m * 32 + 16 + kh * 8 + 2 * j]);
      }
    }
#pragma unroll
    for (int ct = 0; ct < 2; ++ct) {
      int el = cn * 32 + ct * 16 + ln;
      FragAB bf;
#pragma unroll
      for (int j = 0; j < 8; ++j)
        bf.u[j] = ld2(&voT_lds[el * 32 + kh * 8 + 2 * j + (j >= 4 ? 8 : 0)]);
      if (ct == 0) acc0 = wmma_bf16(a, bf, acc0);
      else         acc1 = wmma_bf16(a, bf, acc1);
    }
    __syncthreads();
  }
#pragma unroll
  for (int v = 0; v < 8; ++v) {
    int row = rb * 16 + kh * 8 + v;
    float* op = out + ((size_t)b * T_ + tbase + row) * D_ + ebase;
    op[cn * 32 + ln]      = acc0[v];
    op[cn * 32 + 16 + ln] = acc1[v];
  }
}

// ---------------------------------------------------------------- launch
extern "C" void kernel_launch(void* const* d_in, const int* in_sizes, int n_in,
                              void* d_out, int out_size, void* d_ws,
                              size_t ws_size, hipStream_t stream) {
  const float* x  = (const float*)d_in[0];
  const float* Q  = (const float*)d_in[1];
  const float* K  = (const float*)d_in[2];
  const float* VO = (const float*)d_in[3];
  float* out = (float*)d_out;

  char* ws = (char*)d_ws;
  size_t off = 0;
  auto alloc = [&](size_t bytes) {
    void* p = ws + off;
    off = (off + bytes + 255) & ~(size_t)255;
    return p;
  };
  unsigned short* xb  = (unsigned short*)alloc((size_t)B_ * T_ * D_ * 2);
  unsigned short* Qb  = (unsigned short*)alloc((size_t)H_ * R_ * D_ * 2);
  unsigned short* Kb  = (unsigned short*)alloc((size_t)H_ * R_ * D_ * 2);
  unsigned short* VOb = (unsigned short*)alloc((size_t)H_ * D_ * D_ * 2);
  unsigned short* qws = (unsigned short*)alloc((size_t)B_ * H_ * T_ * R_ * 2);
  unsigned short* kws = (unsigned short*)alloc((size_t)B_ * H_ * T_ * R_ * 2);
  unsigned short* ctx = (unsigned short*)alloc((size_t)B_ * T_ * H_ * D_ * 2);

  int nX = B_ * T_ * D_, nQ = H_ * R_ * D_, nVO = H_ * D_ * D_;
  k_convert<<<(nX + 255) / 256, 256, 0, stream>>>(x, xb, nX);
  k_convert<<<(nQ + 255) / 256, 256, 0, stream>>>(Q, Qb, nQ);
  k_convert<<<(nQ + 255) / 256, 256, 0, stream>>>(K, Kb, nQ);
  k_convert<<<(nVO + 255) / 256, 256, 0, stream>>>(VO, VOb, nVO);

  k_proj<<<B_ * (T_ / 64) * H_, 256, 0, stream>>>(xb, Qb, Kb, qws, kws);
  k_flash<<<B_ * H_ * (T_ / 64), 256, 0, stream>>>(qws, kws, xb, ctx);
  k_out<<<B_ * (T_ / 64) * 4, 256, 0, stream>>>(ctx, VOb, out);
}
